// VQ_15144054686410
// MI455X (gfx1250) — matrix operand order repaired and checked by reading, not compiled, not run on hardware
//
#include <hip/hip_runtime.h>
#include <hip/hip_bf16.h>

// Shapes fixed by the reference: X = [32768, 128] f32, E = [1024, 128] f32.
// Output: q [32768*128] f32 followed by scalar loss at flat index 4194304.

typedef __bf16 bf16_t;
typedef __attribute__((ext_vector_type(16))) __bf16 v16bf;
typedef __attribute__((ext_vector_type(8)))  float  v8f;

#define N_ROWS   32768
#define DIM      128
#define N_CODES  1024

// GEMM staging parameters
#define STAGE_CODES 32          // codes per LDS stage (2 column tiles)
#define NSTAGES     32          // 1024 / 32
#define BROW        136         // padded LDS row stride in bf16 (272 B = 68 dw)
#define BUF_BYTES   17408       // 2 arrays * 32 codes * 272 B

// ---- ws layout (bytes) ----
#define WS_EHI   0          // 1024*128 bf16 = 256 KB
#define WS_ELO   262144     // 256 KB
#define WS_ENORM 524288     // 1024 f32 = 4 KB
#define WS_IDX   528384     // 32768 i32 = 128 KB
#define WS_PART  659456     // 1024 f32 = 4 KB

// ---------------------------------------------------------------------------
// Kernel 1: split embeddings into bf16 hi/lo, precompute ||e||^2 (in f32).
// ---------------------------------------------------------------------------
__global__ __launch_bounds__(256) void vq_prep(const float* __restrict__ E,
                                               bf16_t* __restrict__ Ehi,
                                               bf16_t* __restrict__ Elo,
                                               float* __restrict__ enorm) {
  int k = blockIdx.x * 256 + threadIdx.x;   // 0..1023 (one thread per code)
  const float* er = E + k * DIM;
  float s = 0.0f;
#pragma unroll 4
  for (int d = 0; d < DIM; ++d) {
    float f = er[d];
    s += f * f;
    __bf16 h = (__bf16)f;                 // RNE
    Ehi[k * DIM + d] = h;
    Elo[k * DIM + d] = (__bf16)(f - (float)h);
  }
  enorm[k] = s;
}

// ---------------------------------------------------------------------------
// Kernel 2: split-bf16 WMMA GEMM + fused argmin over 1024 codes.
// 8 waves / block; each wave owns a 16-row tile (A in registers).
// B (E hi/lo) is staged into double-buffered LDS via async global->LDS
// copies (no VGPR round-trip, tracked by ASYNCcnt), shared by all 8 waves.
// ---------------------------------------------------------------------------
__global__ __launch_bounds__(256, 4) void vq_gemm_argmin(
    const float*  __restrict__ X,
    const bf16_t* __restrict__ Ehi,
    const bf16_t* __restrict__ Elo,
    const float*  __restrict__ enorm,
    int*          __restrict__ idx_out) {
  // [buffer][hi/lo][code-in-stage][padded dims] = 2*2*32*136*2 B = 34816 B
  __shared__ __align__(16) bf16_t sB[2][2][STAGE_CODES][BROW];

  const int tid  = threadIdx.x;
  const int wave = tid >> 5;
  const int lane = tid & 31;
  const int lr   = lane & 15;   // row-in-tile (A) / col-in-tile (B,C)
  const int kh   = lane >> 4;   // which K-half this lane holds
  const int rowbase = (blockIdx.x * 8 + wave) * 16;

  // ---- Load A tile row (f32) once, split into bf16 hi/lo in WMMA layout.
  // 16-bit A 16x32 layout: lane holds dims {8*kh..+7} U {16+8*kh..+7}
  // of each K=32 chunk; v16bf element i maps in that order.
  const float* xr = X + (rowbase + lr) * DIM;
  v16bf ahi[4], alo[4];
#pragma unroll
  for (int c = 0; c < 4; ++c) {
    const int k0 = 32 * c + 8 * kh;
#pragma unroll
    for (int j = 0; j < 8; ++j) {
      float f0 = xr[k0 + j];
      float f1 = xr[k0 + 16 + j];
      __bf16 h0 = (__bf16)f0;
      __bf16 h1 = (__bf16)f1;
      ahi[c][j]     = h0;
      ahi[c][8 + j] = h1;
      alo[c][j]     = (__bf16)(f0 - (float)h0);
      alo[c][8 + j] = (__bf16)(f1 - (float)h1);
    }
  }

  // ---- Async staging setup: 16 KB / stage = 1024 x 16B chunks.
  // Chunk c = (wave*4 + i)*32 + lane: arr = c>>9, row = (c>>4)&31, c16 = c&15.
  // One async b128 instruction moves 512 B (16 B per lane); 4 per wave.
  uint32_t ldsb[4];
  const bf16_t* gp[4];
#pragma unroll
  for (int i = 0; i < 4; ++i) {
    const int c   = (wave * 4 + i) * 32 + lane;
    const int arr = c >> 9, rr = (c >> 4) & 31, c16 = c & 15;
    gp[i]   = (arr ? Elo : Ehi) + rr * DIM + c16 * 8;
    // Flat LDS address truncated to 32 bits == LDS byte offset.
    ldsb[i] = (uint32_t)(size_t)&sB[0][arr][rr][c16 * 8];
  }
  auto async_stage = [&](int s, int b) {
#pragma unroll
    for (int i = 0; i < 4; ++i) {
      uint32_t lo = ldsb[i] + b * BUF_BYTES;
      const bf16_t* p = gp[i] + s * (STAGE_CODES * DIM);
      asm volatile("global_load_async_to_lds_b128 %0, %1, off"
                   :: "v"(lo), "v"(p)
                   : "memory");
    }
  };

  // ---- Running argmin of (||e||^2 - 2*S) per C-slot row.
  float minv[8];
  int   mini[8];
#pragma unroll
  for (int r = 0; r < 8; ++r) { minv[r] = 3.4e38f; mini[r] = 0; }

  async_stage(0, 0);
  asm volatile("s_wait_asynccnt 0x0" ::: "memory");
  __syncthreads();

  for (int s = 0; s < NSTAGES; ++s) {
    const int buf = s & 1;
    if (s + 1 < NSTAGES) async_stage(s + 1, buf ^ 1);   // overlap with WMMAs

    // enorm for both column tiles of this stage (overlaps WMMA chain).
    const float en0 = enorm[s * STAGE_CODES + lr];
    const float en1 = enorm[s * STAGE_CODES + 16 + lr];

#pragma unroll
    for (int ctl = 0; ctl < 2; ++ctl) {
      const int cl   = ctl * 16 + lr;          // code within stage
      const int code = s * STAGE_CODES + cl;   // absolute code
      // B 32x16 bf16 layout: lane holds column `code`, dims 32c+{8*kh..+7} U {16+8*kh..+7}.
      const bf16_t* bhp = &sB[buf][0][cl][8 * kh];
      const bf16_t* blp = &sB[buf][1][cl][8 * kh];
      v8f acc = {};
#pragma unroll
      for (int c = 0; c < 4; ++c) {
        union { v16bf v; uint4 u[2]; } bh, bl;   // 16B-aligned LDS reads
        bh.u[0] = *(const uint4*)(bhp + 32 * c);
        bh.u[1] = *(const uint4*)(bhp + 32 * c + 16);
        bl.u[0] = *(const uint4*)(blp + 32 * c);
        bl.u[1] = *(const uint4*)(blp + 32 * c + 16);
        acc = __builtin_amdgcn_wmma_f32_16x16x32_bf16(false, ahi[c], false, bh.v,
                                                      (short)0, acc, false, false);
        acc = __builtin_amdgcn_wmma_f32_16x16x32_bf16(false, alo[c], false, bh.v,
                                                      (short)0, acc, false, false);
        acc = __builtin_amdgcn_wmma_f32_16x16x32_bf16(false, ahi[c], false, bl.v,
                                                      (short)0, acc, false, false);
      }
      const float en = ctl ? en1 : en0;
#pragma unroll
      for (int r = 0; r < 8; ++r) {
        float t = __builtin_fmaf(-2.0f, acc[r], en);   // ||e||^2 - 2*x.e
        if (t < minv[r]) { minv[r] = t; mini[r] = code; }
      }
    }

    if (s + 1 < NSTAGES)
      asm volatile("s_wait_asynccnt 0x0" ::: "memory");  // our fills landed
    __syncthreads();                                     // everyone's fills
  }

  // ---- Butterfly min-reduce across the 16 lanes of each column group.
  // (xor masks 1,2,4,8 never cross the kh boundary at bit 4.)
#pragma unroll
  for (int off = 1; off < 16; off <<= 1) {
#pragma unroll
    for (int r = 0; r < 8; ++r) {
      float ov = __shfl_xor(minv[r], off, 32);
      int   oi = __shfl_xor(mini[r], off, 32);
      if (ov < minv[r] || (ov == minv[r] && oi < mini[r])) {
        minv[r] = ov;
        mini[r] = oi;
      }
    }
  }

  // C layout: VGPR r holds row r (lanes 0-15) / row 8+r (lanes 16-31).
  if (lr < 8) {
    int myidx = mini[0];
#pragma unroll
    for (int r = 1; r < 8; ++r)
      if (lr == r) myidx = mini[r];
    idx_out[rowbase + 8 * kh + lr] = myidx;
  }
}

// ---------------------------------------------------------------------------
// Kernel 3: gather q = E[idx], write output, partial sums of ||q - x||^2.
// ---------------------------------------------------------------------------
__global__ __launch_bounds__(256) void vq_gather(
    const float* __restrict__ X,
    const float* __restrict__ E,
    const int*   __restrict__ idx,
    float*       __restrict__ out,
    float*       __restrict__ partials) {
  const int t = blockIdx.x * 256 + threadIdx.x;   // 262144 threads total
  float s = 0.0f;
  for (int g = t; g < N_ROWS * (DIM / 4); g += 1024 * 256) {
    const int row = g >> 5;            // DIM/4 = 32 float4 per row
    const int d4  = (g & 31) << 2;
    const int id  = idx[row];
    const float4 q = *(const float4*)(E + id * DIM + d4);
    const float4 x = *(const float4*)(X + row * DIM + d4);
    *(float4*)(out + row * DIM + d4) = q;
    const float dx = q.x - x.x, dy = q.y - x.y, dz = q.z - x.z, dw = q.w - x.w;
    s += dx * dx + dy * dy + dz * dz + dw * dw;
  }
#pragma unroll
  for (int off = 16; off; off >>= 1) s += __shfl_xor(s, off, 32);
  __shared__ float sm[8];
  const int wv = threadIdx.x >> 5, ln = threadIdx.x & 31;
  if (ln == 0) sm[wv] = s;
  __syncthreads();
  if (threadIdx.x == 0) {
    float tot = 0.0f;
#pragma unroll
    for (int w = 0; w < 8; ++w) tot += sm[w];
    partials[blockIdx.x] = tot;
  }
}

// ---------------------------------------------------------------------------
// Kernel 4: final reduction; loss = 1.5 * sum  (codebook + 0.5*commitment).
// ---------------------------------------------------------------------------
__global__ __launch_bounds__(256) void vq_finalize(const float* __restrict__ partials,
                                                   float* __restrict__ loss) {
  float s = 0.0f;
  for (int i = threadIdx.x; i < 1024; i += 256) s += partials[i];
#pragma unroll
  for (int off = 16; off; off >>= 1) s += __shfl_xor(s, off, 32);
  __shared__ float sm[8];
  const int wv = threadIdx.x >> 5, ln = threadIdx.x & 31;
  if (ln == 0) sm[wv] = s;
  __syncthreads();
  if (threadIdx.x == 0) {
    float tot = 0.0f;
#pragma unroll
    for (int w = 0; w < 8; ++w) tot += sm[w];
    loss[0] = 1.5f * tot;
  }
}

extern "C" void kernel_launch(void* const* d_in, const int* in_sizes, int n_in,
                              void* d_out, int out_size, void* d_ws, size_t ws_size,
                              hipStream_t stream) {
  const float* X = (const float*)d_in[0];   // 32*32*32*128
  const float* E = (const float*)d_in[1];   // 1024*128
  float* out = (float*)d_out;               // q (4194304) + loss (1)

  char* ws = (char*)d_ws;                   // needs ~664 KB
  bf16_t* Ehi   = (bf16_t*)(ws + WS_EHI);
  bf16_t* Elo   = (bf16_t*)(ws + WS_ELO);
  float*  enorm = (float*)(ws + WS_ENORM);
  int*    idx   = (int*)(ws + WS_IDX);
  float*  part  = (float*)(ws + WS_PART);

  vq_prep<<<4, 256, 0, stream>>>(E, Ehi, Elo, enorm);
  vq_gemm_argmin<<<256, 256, 0, stream>>>(X, Ehi, Elo, enorm, idx);
  vq_gather<<<1024, 256, 0, stream>>>(X, E, idx, out, part);
  vq_finalize<<<1, 256, 0, stream>>>(part, out + N_ROWS * DIM);
}
